// Objective_Func_32495722561842
// MI455X (gfx1250) — compile-verified, matrix operand order checked by reading, not run on hardware
//
#include <hip/hip_runtime.h>
#include <hip/hip_bf16.h>
#include <stdint.h>

typedef __attribute__((ext_vector_type(16))) _Float16 v16h;
typedef __attribute__((ext_vector_type(8)))  float    v8f;
typedef __attribute__((ext_vector_type(4)))  unsigned v4u;
typedef __attribute__((ext_vector_type(8)))  unsigned v8u;

#define NOF_LINKS 64
#define NB        8
#define TLEN      101
#define K100      100
#define DT        0.01f
#define NOISE     0.01f
#define LAM       1.0f
#define RHO       0.1f

// ---------------------------------------------------------------------------
// Kernel 1: per-batch pathloss row-sums via WMMA (P @ ones), then SINR.
//   grid = 512 (one block per b), block = 128 (4 waves, 16 rows per wave).
//   The 64x64 f32 tile (16 KB, contiguous row-major) is staged into LDS by
//   one Tensor Data Mover descriptor (TENSORcnt) issued by wave 0.
// ---------------------------------------------------------------------------
__global__ void sinr_rowsum_kernel(const float* __restrict__ pathloss,
                                   const float* __restrict__ powers,
                                   float* __restrict__ sinr_out) {
    __shared__ float Ptile[NOF_LINKS * NOF_LINKS];   // 16 KB

    const int b    = blockIdx.x;
    const int tid  = threadIdx.x;            // 0..127
    const int lane = tid & 31;
    const int wave = tid >> 5;               // 0..3
    const int i0   = wave * 16;              // row block handled by this wave

    const float* Pb = pathloss + (size_t)b * (NOF_LINKS * NOF_LINKS);

    // --- TDM: one descriptor moves the whole 64x64 f32 tile into LDS ------
    if (wave == 0) {
        const uint64_t ga   = (uint64_t)(uintptr_t)Pb;      // 57-bit global addr
        const unsigned ldsA = (unsigned)(uintptr_t)&Ptile[0];

        v4u g0;                                  // D# group 0 (128b)
        g0[0] = 1u;                              // count=1 (valid), no gather
        g0[1] = ldsA;                            // lds_addr
        g0[2] = (unsigned)ga;                    // global_addr[31:0]
        g0[3] = (unsigned)(ga >> 32) | (2u << 30); // global_addr[56:32] | type=2

        v8u g1;                                  // D# group 1 (256b)
        g1[0] = 2u << 16;                        // wg_mask=0, data_size=4B
        g1[1] = 64u << 16;                       // tensor_dim0 = 64
        g1[2] = 64u << 16;                       // tensor_dim1 = 64
        g1[3] = 64u << 16;                       // tile_dim0   = 64
        g1[4] = 64u;                             // tile_dim1=64, tile_dim2=0
        g1[5] = 64u;                             // tensor_dim0_stride = 64
        g1[6] = 4096u << 16;                     // tensor_dim1_stride = 4096
        g1[7] = 0u;

        v4u gz;                                  // groups 2/3: zeros (2D tile)
        gz[0] = 0u; gz[1] = 0u; gz[2] = 0u; gz[3] = 0u;

        asm volatile("tensor_load_to_lds %0, %1, %2, %3"
                     :: "s"(g0), "s"(g1), "s"(gz), "s"(gz)
                     : "memory");
        __builtin_amdgcn_s_wait_tensorcnt(0);
    }
    __syncthreads();

    // --- build A operands (16x32 f16 tiles of P rows) for two K-chunks ---
    // Row slots for result-row M are lanes M and M+16. With B == ones the
    // K-permutation inside a row is irrelevant; lane L supplies 16 contiguous
    // columns: k0 + (L<16 ? 0 : 16) .. +15 of row i0 + (L&15).
    const int m    = lane & 15;
    const int koff = (lane >> 4) << 4;       // 0 or 16

    v16h a0, a1, ones;
    #pragma unroll
    for (int e = 0; e < 16; ++e) ones[e] = (_Float16)1.0f;

    const float* row0 = &Ptile[(i0 + m) * NOF_LINKS + 0  + koff];
    const float* row1 = &Ptile[(i0 + m) * NOF_LINKS + 32 + koff];
    #pragma unroll
    for (int e = 0; e < 16; ++e) {
        a0[e] = (_Float16)row0[e];
        a1[e] = (_Float16)row1[e];
    }

    v8f c = {};
    // D[M,n] = sum_k A[M,k] * 1  (rowsum replicated across all 16 columns)
    c = __builtin_amdgcn_wmma_f32_16x16x32_f16(false, a0, false, ones,
                                               (short)0, c, false, false);
    c = __builtin_amdgcn_wmma_f32_16x16x32_f16(false, a1, false, ones,
                                               (short)0, c, false, false);

    // --- extract: lanes 0..7 hold rows 0..7 (VGPR v <-> M=v), lanes 16..23
    //     hold rows 8..15 (VGPR v <-> M=8+v). One row per active lane. ---
    const int half = lane >> 4;
    const int l    = lane & 15;
    if (l < 8) {
        const int mrow = half * 8 + l;
        const int i    = i0 + mrow;
        const float rowsum = c[l];
        const float diag   = Ptile[i * NOF_LINKS + i];
        const float p0     = powers[(size_t)b * (NB * NOF_LINKS) + i]; // powers[b,0,i]
        const float rest   = rowsum - diag;
        const float sinr   = (diag * p0) / (p0 * rest + NOISE);
        sinr_out[b * NOF_LINKS + i] = sinr;
    }
}

// ---------------------------------------------------------------------------
// Kernel 2: per-row CDF -> pdf -> 7 truncated self-convolutions (the scan),
//   then fold tails + powers into one objective contribution per row.
//   grid = 32768 (one block per (b,i) row), block = 128.
//   powers[b,:,i] (8 strided f32) are prefetched into LDS with async
//   global->LDS loads (ASYNCcnt) and consumed only after the scan, hiding
//   the full global latency behind ~7*100 FMA iterations.
// ---------------------------------------------------------------------------
__global__ void outage_scan_kernel(const float* __restrict__ powers,
                                   const float* __restrict__ sinr_in,
                                   float* __restrict__ contrib) {
    __shared__ float F[104];       // CDF, 101 used
    __shared__ float fpad[224];    // zero-padded pdf: fpad[99+d] = f[d]
    __shared__ float Qa[104];
    __shared__ float Qb[104];
    __shared__ float tails[NB];
    __shared__ volatile float pvals[NB];   // written only by async DMA

    const int row = blockIdx.x;            // row = b*64 + i
    const int t   = threadIdx.x;           // 0..127
    const int b   = row >> 6;
    const int i   = row & 63;

    // --- async prefetch of powers[b, 0..7, i] into LDS (lanes 0..7) -------
    if (t < NB) {
        const float* base   = powers + (size_t)b * (NB * NOF_LINKS);
        unsigned byteOff    = (unsigned)((t * NOF_LINKS + i) * 4);
        unsigned ldsOff     = (unsigned)(uintptr_t)&pvals[t];
        asm volatile("global_load_async_to_lds_b32 %0, %1, %2 offset:0"
                     :: "v"(ldsOff), "v"(byteOff), "s"(base)
                     : "memory");
    }

    const float sinr = sinr_in[row];
    const float invs = 1.0f / sinr;

    // F(t) = 1 - exp(-(2^t - 1)/sinr),  t = idx * DT, idx = 0..100
    if (t < TLEN) {
        const float p2 = exp2f((float)t * DT) - 1.0f;
        F[t] = 1.0f - expf(-p2 * invs);
    }
    fpad[t] = 0.0f;
    if (t + 128 < 224) fpad[t + 128] = 0.0f;
    __syncthreads();

    if (t < K100) {
        fpad[99 + t] = (F[t + 1] - F[t]) * (1.0f / DT);  // pdf
        Qa[t]        = F[t + 1];                          // Q0
    }
    if (t == 0) tails[0] = F[TLEN - 1];                   // Q0[99]
    __syncthreads();

    float* Q  = Qa;
    float* Qn = Qb;
    for (int n = 1; n < NB; ++n) {
        if (t < K100) {
            // out[t] = sum_{j=0..t} Q[j]*f[t-j]; uniform loop via zero-pad.
            const float* fp = &fpad[99 + t];   // fp[-j] stride-1 across lanes
            float acc = 0.0f;
            #pragma unroll 4
            for (int j = 0; j < K100; ++j) acc += Q[j] * fp[-j];
            const float v = acc * DT;
            Qn[t] = v;
            if (t == K100 - 1) tails[n] = v;
        }
        __syncthreads();
        float* tmp = Q; Q = Qn; Qn = tmp;
    }

    if (t == 0) {
        // powers were issued ~7 convolutions ago; the wait is effectively free.
        asm volatile("s_wait_asynccnt 0x0" ::: "memory");
        float e = pvals[0];                                // p0 term of E
        float dsum = 0.0f;
        #pragma unroll
        for (int n = 1; n < NB; ++n) e += pvals[n] * tails[n - 1];
        #pragma unroll
        for (int n = 0; n < NB - 1; ++n) dsum += tails[n];
        contrib[row] = e + LAM * tails[NB - 1] + RHO * (1.0f + dsum);
    }
}

// ---------------------------------------------------------------------------
// Kernel 3: deterministic fixed-tree reduction of 32768 contributions.
// ---------------------------------------------------------------------------
__global__ void reduce_kernel(const float* __restrict__ contrib,
                              float* __restrict__ out, int n) {
    __shared__ float s[256];
    const int t = threadIdx.x;
    float acc = 0.0f;
    for (int idx = t; idx < n; idx += 256) acc += contrib[idx];
    s[t] = acc;
    __syncthreads();
    #pragma unroll
    for (int w = 128; w > 0; w >>= 1) {
        if (t < w) s[t] += s[t + w];
        __syncthreads();
    }
    if (t == 0) out[0] = s[0];
}

// ---------------------------------------------------------------------------
extern "C" void kernel_launch(void* const* d_in, const int* in_sizes, int n_in,
                              void* d_out, int out_size, void* d_ws, size_t ws_size,
                              hipStream_t stream) {
    const float* pathloss = (const float*)d_in[0];  // [512, 64, 64] f32
    const float* powers   = (const float*)d_in[1];  // [512, 8, 64]  f32
    float* out = (float*)d_out;                     // scalar

    const int B = 512, ROWS = B * NOF_LINKS;        // 32768
    float* ws_sinr    = (float*)d_ws;               // 32768 f32
    float* ws_contrib = ws_sinr + ROWS;             // 32768 f32

    sinr_rowsum_kernel<<<B, 128, 0, stream>>>(pathloss, powers, ws_sinr);
    outage_scan_kernel<<<ROWS, 128, 0, stream>>>(powers, ws_sinr, ws_contrib);
    reduce_kernel<<<1, 256, 0, stream>>>(ws_contrib, out, ROWS);
}
